// ClassifierI_23768349016665
// MI455X (gfx1250) — matrix operand order repaired and checked by reading, not compiled, not run on hardware
//
#include <hip/hip_runtime.h>
#include <hip/hip_bf16.h>
#include <math.h>

// Problem constants (match reference)
#define BB 16
#define TT 512
#define CC 1024
#define HH 16
#define LL 12
#define DD 64
#define MM (BB * TT)   // 8192 rows of tokens

typedef __bf16 bf16;
typedef bf16 bf16x16 __attribute__((ext_vector_type(16)));
typedef bf16 bf16x8  __attribute__((ext_vector_type(8)));
typedef float f32x8  __attribute__((ext_vector_type(8)));
typedef unsigned short u16x8 __attribute__((ext_vector_type(8)));
typedef unsigned int v4u __attribute__((ext_vector_type(4)));
typedef int v8i __attribute__((ext_vector_type(8)));
typedef int v4i __attribute__((ext_vector_type(4)));

#if defined(__has_builtin)
#if __has_builtin(__builtin_amdgcn_tensor_load_to_lds) && \
    __has_builtin(__builtin_amdgcn_s_wait_tensorcnt)
#define USE_TDM 1
#endif
#endif
#ifndef USE_TDM
#define USE_TDM 0
#endif

__device__ __forceinline__ unsigned short f2bf(float f) {
  unsigned int u = __float_as_uint(f);
  u += 0x7FFFu + ((u >> 16) & 1u);          // round-to-nearest-even
  return (unsigned short)(u >> 16);
}
__device__ __forceinline__ float bf2f(unsigned short h) {
  return __uint_as_float(((unsigned int)h) << 16);
}
__device__ __forceinline__ bf16x16 frag_cat(u16x8 lo, u16x8 hi) {
  bf16x8 l = __builtin_bit_cast(bf16x8, lo);
  bf16x8 h = __builtin_bit_cast(bf16x8, hi);
  return __builtin_shufflevector(l, h, 0,1,2,3,4,5,6,7,8,9,10,11,12,13,14,15);
}

#if USE_TDM
// TDM: DMA a (tile_h rows x tile_w bf16) tile, global row stride = stride
// elements, into LDS at byte offset lds_off. LDS padding: after every
// (2<<pad_int_code) DWORDs insert (pad_amt_code+1) DWORDs.
__device__ __forceinline__ void tdm_load_tile(
    const unsigned short* g, unsigned lds_off, int tile_w, int tile_h,
    int stride, unsigned pad_int_code, unsigned pad_amt_code) {
  unsigned long long ga = (unsigned long long)(size_t)g;
  v4u g0;
  g0[0] = 1u;                                  // count=1, user descriptor
  g0[1] = lds_off;                             // lds_addr
  g0[2] = (unsigned)ga;                        // global_addr[31:0]
  g0[3] = (unsigned)((ga >> 32) & 0x01FFFFFFull) | (2u << 30);  // addr[56:32], type=2
  const unsigned td0 = (unsigned)stride;       // tensor_dim0 = row length bound
  v8i g1;
  g1[0] = (int)((1u << 16) | (1u << 20) | (pad_int_code << 22) | (pad_amt_code << 25));
  //            data_size=2B  pad_en
  g1[1] = (int)((td0 & 0xFFFFu) << 16);        // atomic_addr=0 | tensor_dim0[15:0]
  g1[2] = (int)(td0 >> 16);                    // tensor_dim0[31:16] | tensor_dim1[15:0]=0
  g1[3] = (int)(0x10u | ((unsigned)tile_w << 16)); // tensor_dim1[31:16]=16 (1M) | tile_dim0
  g1[4] = (int)(unsigned)tile_h;               // tile_dim1 | tile_dim2=0
  g1[5] = (int)td0;                            // tensor_dim0_stride[31:0]
  g1[6] = 0;                                   // stride[47:32]=0 | dim1_stride[15:0]=0
  g1[7] = 0;
  v4i g2 = {0, 0, 0, 0};                       // 2D tensor: groups 2/3 unused
  v4i g3 = {0, 0, 0, 0};
#if __clang_major__ >= 23
  v8i g4 = {0, 0, 0, 0, 0, 0, 0, 0};
  __builtin_amdgcn_tensor_load_to_lds(g0, g1, g2, g3, g4, 0);
#else
  __builtin_amdgcn_tensor_load_to_lds(g0, g1, g2, g3, 0);
#endif
}
#endif

// ---------------------------------------------------------------------------
// Weight transpose + fp32 -> bf16: out[l][N][K] = in[l][K][N]^T
// ---------------------------------------------------------------------------
__global__ __launch_bounds__(256) void transpose_to_bf16_kernel(
    const float* __restrict__ in, unsigned short* __restrict__ out,
    int K, int N) {
  __shared__ float tile[32][33];
  const int l = blockIdx.z;
  const float* src = in + (size_t)l * K * N;
  unsigned short* dst = out + (size_t)l * N * K;
  const int n0 = blockIdx.x * 32, k0 = blockIdx.y * 32;
  const int tx = threadIdx.x & 31, ty = threadIdx.x >> 5;
  #pragma unroll
  for (int j = 0; j < 4; j++)
    tile[ty + j * 8][tx] = src[(size_t)(k0 + ty + j * 8) * N + n0 + tx];
  __syncthreads();
  #pragma unroll
  for (int j = 0; j < 4; j++)
    dst[(size_t)(n0 + ty + j * 8) * K + k0 + tx] = f2bf(tile[tx][ty + j * 8]);
}

// ---------------------------------------------------------------------------
// Embedding: x[b,t,c] = wte[idx[b,t],c] + wpe[t,c]
// ---------------------------------------------------------------------------
__global__ __launch_bounds__(256) void embed_kernel(
    const int* __restrict__ idx, const float* __restrict__ wte,
    const float* __restrict__ wpe, float* __restrict__ x) {
  size_t i = (size_t)blockIdx.x * 256 + threadIdx.x;  // over MM*CC exactly
  int c = (int)(i % CC);
  size_t bt = i / CC;
  int t = (int)(bt % TT);
  int tok = idx[bt];
  x[i] = wte[(size_t)tok * CC + c] + wpe[(size_t)t * CC + c];
}

// ---------------------------------------------------------------------------
// LayerNorm over C=1024, one block per token, bf16 output
// ---------------------------------------------------------------------------
__global__ __launch_bounds__(256) void ln_kernel(
    const float* __restrict__ x, const float* __restrict__ w,
    const float* __restrict__ b, unsigned short* __restrict__ out) {
  __shared__ float red[256];
  const int tok = blockIdx.x;
  const float* xr = x + (size_t)tok * CC;

  float s = 0.f;
  for (int c = threadIdx.x; c < CC; c += 256) s += xr[c];
  red[threadIdx.x] = s; __syncthreads();
  for (int o = 128; o > 0; o >>= 1) {
    if (threadIdx.x < o) red[threadIdx.x] += red[threadIdx.x + o];
    __syncthreads();
  }
  const float mean = red[0] * (1.0f / CC);
  __syncthreads();

  float vs = 0.f;
  for (int c = threadIdx.x; c < CC; c += 256) { float d = xr[c] - mean; vs += d * d; }
  red[threadIdx.x] = vs; __syncthreads();
  for (int o = 128; o > 0; o >>= 1) {
    if (threadIdx.x < o) red[threadIdx.x] += red[threadIdx.x + o];
    __syncthreads();
  }
  const float rstd = rsqrtf(red[0] * (1.0f / CC) + 1e-5f);

  unsigned short* orow = out + (size_t)tok * CC;
  for (int c = threadIdx.x; c < CC; c += 256)
    orow[c] = f2bf((xr[c] - mean) * rstd * w[c] + b[c]);
}

// Fragment loads + 8 WMMA for one 32-element K-step from staged LDS tiles.
constexpr int LDT = 40;  // LDS row stride in ushorts (80B: 16B-aligned rows)
__device__ __forceinline__ void gemm_compute_step(
    const unsigned short* lAp, const unsigned short* lBp,
    f32x8 (&acc)[4][2], int wm, int wn, int half, int l15) {
  bf16x16 afr[4];
  #pragma unroll
  for (int mi = 0; mi < 4; mi++) {
    const unsigned short* p = &lAp[(wm + mi * 16 + l15) * LDT];
    afr[mi] = frag_cat(*(const u16x8*)(p + 8 * half),
                       *(const u16x8*)(p + 16 + 8 * half));
  }
  bf16x16 bfr[2];
  #pragma unroll
  for (int ni = 0; ni < 2; ni++) {
    const unsigned short* p = &lBp[(wn + ni * 16 + l15) * LDT];
    bfr[ni] = frag_cat(*(const u16x8*)(p + 8 * half), *(const u16x8*)(p + 16 + 8 * half));
  }
  #pragma unroll
  for (int mi = 0; mi < 4; mi++)
    #pragma unroll
    for (int ni = 0; ni < 2; ni++)
      acc[mi][ni] = __builtin_amdgcn_wmma_f32_16x16x32_bf16(
          false, afr[mi], false, bfr[ni], (short)0, acc[mi][ni], false, false);
}

// ---------------------------------------------------------------------------
// bf16 WMMA GEMM: D = A[M,K] * Bt[N,K]^T (+bias +residual, opt GELU)
// Weights pre-transposed to [N,K]: A and B staging are identical row-major
// 128x32 tile copies. TDM path: double-buffered LDS, wave 0 DMAs the next
// tile while all waves compute the current one (TENSORcnt pipeline).
// ---------------------------------------------------------------------------
template<bool RES, bool GELU, bool W32, bool WBF>
__global__ __launch_bounds__(256) void gemm_bf16_kernel(
    const unsigned short* __restrict__ A,    // [M,K] bf16
    const unsigned short* __restrict__ Bt,   // [N,K] bf16 (transposed weights)
    const float* __restrict__ bias,          // [N]
    const float* __restrict__ resid,         // [M,N] fp32 (if RES)
    float* __restrict__ out32,               // [M,N] fp32 (if W32)
    unsigned short* __restrict__ outbf,      // [M,N] bf16 (if WBF)
    int M, int N, int K) {
#if USE_TDM
  __shared__ unsigned short lA[2][128 * LDT];
  __shared__ unsigned short lB[2][128 * LDT];
#else
  __shared__ unsigned short lA[1][128 * LDT];
  __shared__ unsigned short lB[1][128 * LDT];
#endif

  const int tid  = threadIdx.x;
  const int lane = tid & 31;
  const int wave = tid >> 5;
  const int half = lane >> 4;
  const int l15  = lane & 15;

  const int m0 = blockIdx.y * 128;
  const int n0 = blockIdx.x * 128;
  const int wm = (wave & 1) * 64;
  const int wn = (wave >> 1) * 32;

  f32x8 acc[4][2];
  #pragma unroll
  for (int i = 0; i < 4; i++)
    #pragma unroll
    for (int j = 0; j < 2; j++)
      #pragma unroll
      for (int e = 0; e < 8; e++) acc[i][j][e] = 0.0f;

  const int row = tid >> 1;          // 0..127 staging/prefetch row
  const int kh  = (tid & 1) * 16;    // k half within 32

#if USE_TDM
  const unsigned offA[2] = { (unsigned)(size_t)(void*)&lA[0][0],
                             (unsigned)(size_t)(void*)&lA[1][0] };
  const unsigned offB[2] = { (unsigned)(size_t)(void*)&lB[0][0],
                             (unsigned)(size_t)(void*)&lB[1][0] };
  // prologue: DMA tile 0 into buffer 0
  if (wave == 0) {
    tdm_load_tile(A  + (size_t)m0 * K, offA[0], 32, 128, K, 3, 3);
    tdm_load_tile(Bt + (size_t)n0 * K, offB[0], 32, 128, K, 3, 3);
    __builtin_amdgcn_s_wait_tensorcnt(0);
  }
  __syncthreads();

  for (int k0 = 0; k0 < K; k0 += 32) {
    const int cur = (k0 >> 5) & 1;
    // kick off DMA for the next tile into the other buffer (consumed last
    // iteration; end-of-iteration barrier ordered those reads before this)
    if (k0 + 32 < K && wave == 0) {
      tdm_load_tile(A  + (size_t)m0 * K + k0 + 32, offA[cur ^ 1], 32, 128, K, 3, 3);
      tdm_load_tile(Bt + (size_t)n0 * K + k0 + 32, offB[cur ^ 1], 32, 128, K, 3, 3);
    }
    // prefetch two steps ahead into GL2 (next step is already in flight)
    if (k0 + 64 < K) {
      __builtin_prefetch(A  + (size_t)(m0 + row) * K + k0 + 64 + kh, 0, 3);
      __builtin_prefetch(Bt + (size_t)(n0 + row) * K + k0 + 64 + kh, 0, 3);
    }
    gemm_compute_step(lA[cur], lB[cur], acc, wm, wn, half, l15);
    // drain DMA before anyone touches the freshly written buffer
    if (wave == 0) __builtin_amdgcn_s_wait_tensorcnt(0);
    __syncthreads();
  }
#else
  for (int k0 = 0; k0 < K; k0 += 32) {
    __syncthreads();
    {
      const unsigned short* ga = A + (size_t)(m0 + row) * K + k0 + kh;
      *(u16x8*)(&lA[0][row * LDT + kh])     = *(const u16x8*)(ga);
      *(u16x8*)(&lA[0][row * LDT + kh + 8]) = *(const u16x8*)(ga + 8);
      const unsigned short* gb = Bt + (size_t)(n0 + row) * K + k0 + kh;
      *(u16x8*)(&lB[0][row * LDT + kh])     = *(const u16x8*)(gb);
      *(u16x8*)(&lB[0][row * LDT + kh + 8]) = *(const u16x8*)(gb + 8);
    }
    if (k0 + 32 < K) {
      __builtin_prefetch(A  + (size_t)(m0 + row) * K + k0 + 32 + kh, 0, 3);
      __builtin_prefetch(Bt + (size_t)(n0 + row) * K + k0 + 32 + kh, 0, 3);
    }
    __syncthreads();
    gemm_compute_step(lA[0], lB[0], acc, wm, wn, half, l15);
  }
#endif

  // Epilogue: C/D layout: vgpr r -> M = r + 8*half, lane l15 -> N
  #pragma unroll
  for (int mi = 0; mi < 4; mi++) {
    #pragma unroll
    for (int ni = 0; ni < 2; ni++) {
      #pragma unroll
      for (int r = 0; r < 8; r++) {
        const int m = m0 + wm + mi * 16 + r + 8 * half;
        const int n = n0 + wn + ni * 16 + l15;
        float v = acc[mi][ni][r] + bias[n];
        if (RES) v += resid[(size_t)m * N + n];
        if (GELU) {
          float xg = v;
          v = 0.5f * xg * (1.0f + tanhf(0.7978845608028654f *
                                        (xg + 0.044715f * xg * xg * xg)));
        }
        if (W32) out32[(size_t)m * N + n] = v;
        if (WBF) outbf[(size_t)m * N + n] = f2bf(v);
      }
    }
  }
}

// ---------------------------------------------------------------------------
// Flash attention with WMMA. One block = (b, h, 128 q-rows); 8 waves, each
// owns 16 q rows. K tile staged by TDM (row stride 3C, padded to 72 ushorts);
// V staged manually (needs transpose). Per 32-key tile: 4 wmma for S,
// online softmax, P via wave-private LDS transpose, 4 wmma for O.
// ---------------------------------------------------------------------------
__global__ __launch_bounds__(256) void attn_flash_kernel(
    const unsigned short* __restrict__ qkv, unsigned short* __restrict__ y) {
  constexpr int LKD = 72;
  constexpr int LVD = 40;
  constexpr int LPD = 40;
  __shared__ unsigned short lK[32 * LKD];       // K tile [t][d]  32x64
  __shared__ unsigned short lV[64 * LVD];       // V tile [d][t]  64x32 (transposed)
  __shared__ unsigned short lP[8 * 16 * LPD];   // per-wave P scratch 16x32

  const int tid  = threadIdx.x;
  const int lane = tid & 31;
  const int wave = tid >> 5;
  const int half = lane >> 4;
  const int l15  = lane & 15;

  const int bh = blockIdx.y;
  const int b  = bh >> 4;
  const int h  = bh & (HH - 1);
  const int q0 = blockIdx.x * 128;
  const int qrow = q0 + wave * 16;

  const size_t rs = 3 * CC;
  const unsigned short* base = qkv + (size_t)b * TT * rs + h * DD;
#if USE_TDM
  const unsigned offK = (unsigned)(size_t)(void*)&lK[0];
#endif

  bf16x16 qf[2];
  {
    const unsigned short* qr = base + (size_t)(qrow + l15) * rs;
    #pragma unroll
    for (int ks = 0; ks < 2; ks++)
      qf[ks] = frag_cat(*(const u16x8*)(qr + 32 * ks + 8 * half),
                        *(const u16x8*)(qr + 32 * ks + 16 + 8 * half));
  }

  f32x8 o[4];
  #pragma unroll
  for (int dt = 0; dt < 4; dt++)
    #pragma unroll
    for (int e = 0; e < 8; e++) o[dt][e] = 0.0f;
  float mrun[8], lrun[8];
  #pragma unroll
  for (int r = 0; r < 8; r++) { mrun[r] = -3.0e38f; lrun[r] = 0.0f; }

  const int ktiles = (q0 + 128) / 32;
  for (int kt = 0; kt < ktiles; kt++) {
    const int t0 = kt * 32;
    __syncthreads();
#if USE_TDM
    if (wave == 0)   // K tile: 32 rows x 64 elems, pad 128B rows by 16B -> 72us
      tdm_load_tile(base + (size_t)t0 * rs + CC, offK, 64, 32, (int)rs, 4, 3);
#endif
    {
      const int t  = tid >> 3;
      const int dc = (tid & 7) * 8;
      const unsigned short* gk = base + (size_t)(t0 + t) * rs + CC + dc;
#if !USE_TDM
      *(u16x8*)(&lK[t * LKD + dc]) = *(const u16x8*)(gk);
#endif
      u16x8 vv = *(const u16x8*)(gk + CC);
      #pragma unroll
      for (int i = 0; i < 8; i++) lV[(dc + i) * LVD + t] = vv[i];
    }
#if USE_TDM
    if (wave == 0) __builtin_amdgcn_s_wait_tensorcnt(0);
#endif
    __syncthreads();

    f32x8 s[2];
    #pragma unroll
    for (int nt = 0; nt < 2; nt++) {
      #pragma unroll
      for (int e = 0; e < 8; e++) s[nt][e] = 0.0f;
      #pragma unroll
      for (int ks = 0; ks < 2; ks++) {
        const unsigned short* p = &lK[(nt * 16 + l15) * LKD + 32 * ks];
        bf16x16 kf = frag_cat(*(const u16x8*)(p + 8 * half), *(const u16x8*)(p + 16 + 8 * half));
        s[nt] = __builtin_amdgcn_wmma_f32_16x16x32_bf16(
            false, qf[ks], false, kf, (short)0, s[nt], false, false);
      }
    }

    unsigned short* pw = &lP[wave * 16 * LPD];
    #pragma unroll
    for (int r = 0; r < 8; r++) {
      const int qq = qrow + r + 8 * half;
      float s0 = s[0][r] * 0.125f;
      float s1 = s[1][r] * 0.125f;
      if (t0 + l15 > qq)      s0 = -3.0e38f;
      if (t0 + 16 + l15 > qq) s1 = -3.0e38f;
      float mx = fmaxf(s0, s1);
      mx = fmaxf(mx, __shfl_xor(mx, 1, 32));
      mx = fmaxf(mx, __shfl_xor(mx, 2, 32));
      mx = fmaxf(mx, __shfl_xor(mx, 4, 32));
      mx = fmaxf(mx, __shfl_xor(mx, 8, 32));
      const float mnew = fmaxf(mrun[r], mx);
      const float corr = __expf(mrun[r] - mnew);
      const float p0 = __expf(s0 - mnew);
      const float p1 = __expf(s1 - mnew);
      float rsum = p0 + p1;
      rsum += __shfl_xor(rsum, 1, 32);
      rsum += __shfl_xor(rsum, 2, 32);
      rsum += __shfl_xor(rsum, 4, 32);
      rsum += __shfl_xor(rsum, 8, 32);
      lrun[r] = lrun[r] * corr + rsum;
      mrun[r] = mnew;
      #pragma unroll
      for (int dt = 0; dt < 4; dt++) o[dt][r] *= corr;
      pw[(r + 8 * half) * LPD + l15]      = f2bf(p0);
      pw[(r + 8 * half) * LPD + 16 + l15] = f2bf(p1);
    }

    const unsigned short* pr = &lP[wave * 16 * LPD + l15 * LPD];
    bf16x16 pf = frag_cat(*(const u16x8*)(pr + 8 * half),
                          *(const u16x8*)(pr + 16 + 8 * half));

    #pragma unroll
    for (int dt = 0; dt < 4; dt++) {
      const unsigned short* p = &lV[(dt * 16 + l15) * LVD];
      bf16x16 vf = frag_cat(*(const u16x8*)(p + 8 * half), *(const u16x8*)(p + 16 + 8 * half));
      o[dt] = __builtin_amdgcn_wmma_f32_16x16x32_bf16(
          false, pf, false, vf, (short)0, o[dt], false, false);
    }
  }

  unsigned short* yb = y + (size_t)b * TT * CC + h * DD;
  #pragma unroll
  for (int r = 0; r < 8; r++) {
    const float inv = 1.0f / lrun[r];
    const int qq = qrow + r + 8 * half;
    unsigned short* yr = yb + (size_t)qq * CC;
    #pragma unroll
    for (int dt = 0; dt < 4; dt++)
      yr[dt * 16 + l15] = f2bf(o[dt][r] * inv);
  }
}

// ---------------------------------------------------------------------------
// Head: logits[b,j] = mean_t 3*tanh(0.3*(hf[b,t,:]@head_w[:,j] + head_b[j]))
// ---------------------------------------------------------------------------
__global__ __launch_bounds__(256) void head_kernel(
    const unsigned short* __restrict__ hf, const float* __restrict__ hw,
    const float* __restrict__ hb, float* __restrict__ out) {
  __shared__ float r0[256], r1[256];
  const int b = blockIdx.x;
  float s0 = 0.f, s1 = 0.f;
  for (int t = threadIdx.x; t < TT; t += 256) {
    const unsigned short* xr = hf + ((size_t)b * TT + t) * CC;
    float z0 = hb[0], z1 = hb[1];
    for (int c = 0; c < CC; c++) {
      const float xv = bf2f(xr[c]);
      z0 += xv * hw[c * 2 + 0];
      z1 += xv * hw[c * 2 + 1];
    }
    s0 += 3.0f * tanhf(0.3f * z0);
    s1 += 3.0f * tanhf(0.3f * z1);
  }
  r0[threadIdx.x] = s0; r1[threadIdx.x] = s1; __syncthreads();
  for (int o = 128; o > 0; o >>= 1) {
    if (threadIdx.x < o) { r0[threadIdx.x] += r0[threadIdx.x + o];
                           r1[threadIdx.x] += r1[threadIdx.x + o]; }
    __syncthreads();
  }
  if (threadIdx.x == 0) {
    out[b * 2 + 0] = r0[0] * (1.0f / TT);
    out[b * 2 + 1] = r1[0] * (1.0f / TT);
  }
}

// ---------------------------------------------------------------------------
// Host orchestration
// ---------------------------------------------------------------------------
extern "C" void kernel_launch(void* const* d_in, const int* in_sizes, int n_in,
                              void* d_out, int out_size, void* d_ws, size_t ws_size,
                              hipStream_t stream) {
  (void)in_sizes; (void)n_in; (void)out_size; (void)ws_size;

  const int*   idx    = (const int*)  d_in[0];
  const float* wte    = (const float*)d_in[1];
  const float* wpe    = (const float*)d_in[2];
  const float* ln1_w  = (const float*)d_in[3];
  const float* ln1_b  = (const float*)d_in[4];
  const float* attn_w = (const float*)d_in[5];
  const float* attn_b = (const float*)d_in[6];
  const float* proj_w = (const float*)d_in[7];
  const float* proj_b = (const float*)d_in[8];
  const float* ln2_w  = (const float*)d_in[9];
  const float* ln2_b  = (const float*)d_in[10];
  const float* fc_w   = (const float*)d_in[11];
  const float* fc_b   = (const float*)d_in[12];
  const float* fcp_w  = (const float*)d_in[13];
  const float* fcp_b  = (const float*)d_in[14];
  const float* lnf_w  = (const float*)d_in[15];
  const float* lnf_b  = (const float*)d_in[16];
  const float* head_w = (const float*)d_in[17];
  const float* head_b = (const float*)d_in[18];
  float* out = (float*)d_out;

  // Workspace carve-out
  char* ws = (char*)d_ws;
  size_t off = 0;
  auto alloc = [&](size_t bytes) -> char* {
    char* p = ws + off;
    off = (off + bytes + 255) & ~(size_t)255;
    return p;
  };
  const size_t nWq = (size_t)LL * CC * 3 * CC;
  const size_t nWp = (size_t)LL * CC * CC;
  const size_t nWf = (size_t)LL * CC * 4 * CC;
  const size_t nWj = (size_t)LL * 4 * CC * CC;
  unsigned short* wqkv = (unsigned short*)alloc(nWq * 2);  // [l][3C][C] transposed
  unsigned short* wprj = (unsigned short*)alloc(nWp * 2);  // [l][C][C]
  unsigned short* wfc  = (unsigned short*)alloc(nWf * 2);  // [l][4C][C]
  unsigned short* wfcp = (unsigned short*)alloc(nWj * 2);  // [l][C][4C]
  float*          x    = (float*)         alloc((size_t)MM * CC * 4);
  unsigned short* hbuf = (unsigned short*)alloc((size_t)MM * CC * 2);
  unsigned short* qkvb = (unsigned short*)alloc((size_t)MM * 3 * CC * 2);
  unsigned short* ybuf = (unsigned short*)alloc((size_t)MM * CC * 2);
  unsigned short* gbuf = (unsigned short*)alloc((size_t)MM * 4 * CC * 2);

  // Weight transpose+convert: [K,N] fp32 -> [N,K] bf16 (per layer)
  transpose_to_bf16_kernel<<<dim3(3 * CC / 32, CC / 32, LL), 256, 0, stream>>>(
      attn_w, wqkv, CC, 3 * CC);
  transpose_to_bf16_kernel<<<dim3(CC / 32, CC / 32, LL), 256, 0, stream>>>(
      proj_w, wprj, CC, CC);
  transpose_to_bf16_kernel<<<dim3(4 * CC / 32, CC / 32, LL), 256, 0, stream>>>(
      fc_w, wfc, CC, 4 * CC);
  transpose_to_bf16_kernel<<<dim3(CC / 32, 4 * CC / 32, LL), 256, 0, stream>>>(
      fcp_w, wfcp, 4 * CC, CC);

  // Embedding
  embed_kernel<<<(unsigned)(((size_t)MM * CC) / 256), 256, 0, stream>>>(idx, wte, wpe, x);

  const dim3 blk(256);
  for (int l = 0; l < LL; l++) {
    const unsigned short* Wq = wqkv + (size_t)l * CC * 3 * CC;
    const unsigned short* Wp = wprj + (size_t)l * CC * CC;
    const unsigned short* Wf = wfc  + (size_t)l * CC * 4 * CC;
    const unsigned short* Wj = wfcp + (size_t)l * 4 * CC * CC;

    // ln1 -> h
    ln_kernel<<<MM, blk, 0, stream>>>(x, ln1_w + (size_t)l * CC, ln1_b + (size_t)l * CC, hbuf);
    // qkv = h @ Wq^T + b (bf16 out)
    gemm_bf16_kernel<false, false, false, true>
        <<<dim3(3 * CC / 128, MM / 128), blk, 0, stream>>>(
            hbuf, Wq, attn_b + (size_t)l * 3 * CC, nullptr, nullptr, qkvb,
            MM, 3 * CC, CC);
    // flash attention -> y (bf16)
    attn_flash_kernel<<<dim3(TT / 128, BB * HH), blk, 0, stream>>>(qkvb, ybuf);
    // x = x + y @ Wp^T + b
    gemm_bf16_kernel<true, false, true, false>
        <<<dim3(CC / 128, MM / 128), blk, 0, stream>>>(
            ybuf, Wp, proj_b + (size_t)l * CC, x, x, nullptr, MM, CC, CC);
    // ln2 -> h
    ln_kernel<<<MM, blk, 0, stream>>>(x, ln2_w + (size_t)l * CC, ln2_b + (size_t)l * CC, hbuf);
    // g = gelu(h @ Wf^T + b) (bf16 out)
    gemm_bf16_kernel<false, true, false, true>
        <<<dim3(4 * CC / 128, MM / 128), blk, 0, stream>>>(
            hbuf, Wf, fc_b + (size_t)l * 4 * CC, nullptr, nullptr, gbuf,
            MM, 4 * CC, CC);
    // x = x + g @ Wj^T + b
    gemm_bf16_kernel<true, false, true, false>
        <<<dim3(CC / 128, MM / 128), blk, 0, stream>>>(
            gbuf, Wj, fcp_b + (size_t)l * CC, x, x, nullptr, MM, CC, 4 * CC);
  }

  // Final LN -> hbuf, then head
  ln_kernel<<<MM, blk, 0, stream>>>(x, lnf_w, lnf_b, hbuf);
  head_kernel<<<BB, blk, 0, stream>>>(hbuf, head_w, head_b, out);
}
